// Seq2SeqLSTM_74878459838799
// MI455X (gfx1250) — compile-verified, matrix operand order checked by reading, not run on hardware
//
#include <hip/hip_runtime.h>

// ---------------- problem constants ----------------
#define IC    64      // input size
#define HC    512     // hidden size
#define GC    2048    // 4*HC gate columns
#define KC    576     // IC + HC fused K dimension
#define TENC  256
#define TDEC  10
#define BTOT  2048
#define BM    16      // batch rows per workgroup
#define LDST  584     // LDS row stride in bf16 (16B-aligned, bank-conflict-free)

typedef __bf16 v16bf __attribute__((ext_vector_type(16)));
typedef __bf16 v8bf  __attribute__((ext_vector_type(8)));
typedef float  v8f   __attribute__((ext_vector_type(8)));

union Frag16 { v16bf v; v8bf h[2]; };

__device__ __forceinline__ float sigm_(float x) {
    return 1.0f / (1.0f + __expf(-x));
}
__device__ __forceinline__ float tanh_(float x) {
    // tanh(x) = 1 - 2/(exp(2x)+1)  (fast v_exp path)
    return 1.0f - 2.0f / (__expf(2.0f * x) + 1.0f);
}

// Pack [Wih | Whh] (GC x KC) into bf16 row-major, and bias = bih + bhh.
__global__ void s2s_prep_kernel(const float* __restrict__ Wih,
                                const float* __restrict__ Whh,
                                const float* __restrict__ bih,
                                const float* __restrict__ bhh,
                                __bf16* __restrict__ Wout,
                                float* __restrict__ bout) {
    int idx = blockIdx.x * 256 + threadIdx.x;
    if (idx < GC) bout[idx] = bih[idx] + bhh[idx];
    if (idx >= GC * KC) return;
    int n = idx / KC;
    int k = idx - n * KC;
    float v = (k < IC) ? Wih[n * IC + k] : Whh[n * HC + (k - IC)];
    Wout[idx] = (__bf16)v;
}

__global__ __launch_bounds__(256)
void s2s_lstm_kernel(const float* __restrict__ x,
                     const __bf16* __restrict__ Wenc, const float* __restrict__ benc,
                     const __bf16* __restrict__ Wdec, const float* __restrict__ bdec,
                     const float* __restrict__ linW, const float* __restrict__ linb,
                     const float* __restrict__ outW, const float* __restrict__ outb,
                     float* __restrict__ out) {
    __shared__ __align__(16) __bf16 Asb[BM][LDST];   // [x_t | h] tile, bf16
    __shared__ __align__(16) float  decb[BM][IC];    // decoder fp32 staging

    const int tid  = threadIdx.x;
    const int wave = tid >> 5;
    const int lane = tid & 31;
    const int half = lane >> 4;     // 0: K {0..7,16..23}, 1: K {8..15,24..31}
    const int l    = lane & 15;     // row (A) / column (B) within tile
    const int b0   = blockIdx.x * BM;

    // zero-init LDS (h0 = 0); x region is overwritten each step anyway
    for (int i = tid; i < BM * LDST; i += 256)
        Asb[i / LDST][i % LDST] = (__bf16)0.0f;

    // cell state c, register-resident across all timesteps
    v8f cst[4];
    #pragma unroll
    for (int jt = 0; jt < 4; ++jt)
        #pragma unroll
        for (int j = 0; j < 8; ++j) cst[jt][j] = 0.0f;

    const int koff = half * 8;

    // One LSTM cell: gates = A(16xKC) @ W^T via WMMA, then elementwise update.
    auto cell = [&](const __bf16* __restrict__ Win, const float* __restrict__ bias) {
        v8f acc[4][4];
        #pragma unroll
        for (int g = 0; g < 4; ++g)
            #pragma unroll
            for (int jt = 0; jt < 4; ++jt)
                #pragma unroll
                for (int j = 0; j < 8; ++j) acc[g][jt][j] = 0.0f;

        // Opaque per-lane BYTE offset: blocks LICM from hoisting the
        // (t-invariant) weight loads out of the timestep loop, while the base
        // pointer keeps its global address space -> global_load_b128 (GVS),
        // not flat_load (which would couple the stream to DScnt).
        unsigned wbyte = ((unsigned)(wave * 64 + l) * KC + (unsigned)koff) * 2u;
        asm volatile("" : "+v"(wbyte));
        const __bf16* wbase = (const __bf16*)((const char*)Win + wbyte);

        #pragma unroll 1
        for (int kc = 0; kc < KC; kc += 32) {
            Frag16 a;
            a.h[0] = *(const v8bf*)&Asb[l][kc + koff];
            a.h[1] = *(const v8bf*)&Asb[l][kc + koff + 16];
            const __bf16* wk = wbase + kc;
            #pragma unroll
            for (int g = 0; g < 4; ++g) {
                #pragma unroll
                for (int jt = 0; jt < 4; ++jt) {
                    Frag16 b;
                    b.h[0] = *(const v8bf*)(wk + (size_t)(g * HC + jt * 16) * KC);
                    b.h[1] = *(const v8bf*)(wk + (size_t)(g * HC + jt * 16) * KC + 16);
                    acc[g][jt] = __builtin_amdgcn_wmma_f32_16x16x32_bf16(
                        false, a.v, false, b.v, (short)0, acc[g][jt], false, false);
                }
            }
        }

        // all waves must finish reading A (h region) before we overwrite h
        __syncthreads();

        #pragma unroll
        for (int jt = 0; jt < 4; ++jt) {
            const int ncol = wave * 64 + jt * 16 + l;   // hidden unit index
            const float bi = bias[0 * HC + ncol];
            const float bf = bias[1 * HC + ncol];
            const float bg = bias[2 * HC + ncol];
            const float bo = bias[3 * HC + ncol];
            #pragma unroll
            for (int j = 0; j < 8; ++j) {
                const float iv = sigm_(acc[0][jt][j] + bi);
                const float fv = sigm_(acc[1][jt][j] + bf);
                const float gv = tanh_(acc[2][jt][j] + bg);
                const float ov = sigm_(acc[3][jt][j] + bo);
                const float cv = fv * cst[jt][j] + iv * gv;
                cst[jt][j] = cv;
                const float hv = ov * tanh_(cv);
                Asb[j + half * 8][IC + ncol] = (__bf16)hv;   // publish h for next step
            }
        }
    };

    // ---------------- encoder: 256 steps ----------------
    #pragma unroll 1
    for (int t = 0; t < TENC; ++t) {
        __syncthreads();   // prev h writes done; LDS x region free
        {
            const int row = tid >> 4;          // 0..15
            const int cg  = (tid & 15) << 2;   // 0..60
            const float4 xv =
                *(const float4*)(x + ((size_t)(b0 + row) * TENC + t) * IC + cg);
            Asb[row][cg + 0] = (__bf16)xv.x;
            Asb[row][cg + 1] = (__bf16)xv.y;
            Asb[row][cg + 2] = (__bf16)xv.z;
            Asb[row][cg + 3] = (__bf16)xv.w;
        }
        __syncthreads();
        cell(Wenc, benc);
    }
    // After the loop, Asb cols [0,64) hold x[:,255,:] == dec_in0, h is final enc state.

    // ---------------- decoder: 10 steps ----------------
    #pragma unroll 1
    for (int s = 0; s < TDEC; ++s) {
        __syncthreads();          // input (cols 0..63) + h visible
        cell(Wdec, bdec);
        __syncthreads();          // new h visible

        // dec_out = h @ lin_dec_W^T + lin_dec_b   (16 x 64, K=512)
        {
            const int row   = tid & 15;
            const int cbase = (tid >> 4) << 2;   // 0..60
            float s0 = linb[cbase + 0], s1 = linb[cbase + 1];
            float s2 = linb[cbase + 2], s3 = linb[cbase + 3];
            const float* w0 = linW + (size_t)(cbase + 0) * HC;
            const float* w1 = linW + (size_t)(cbase + 1) * HC;
            const float* w2 = linW + (size_t)(cbase + 2) * HC;
            const float* w3 = linW + (size_t)(cbase + 3) * HC;
            for (int kk = 0; kk < HC; ++kk) {
                const float hv = (float)Asb[row][IC + kk];
                s0 += hv * w0[kk];
                s1 += hv * w1[kk];
                s2 += hv * w2[kk];
                s3 += hv * w3[kk];
            }
            decb[row][cbase + 0] = s0;  Asb[row][cbase + 0] = (__bf16)s0;
            decb[row][cbase + 1] = s1;  Asb[row][cbase + 1] = (__bf16)s1;
            decb[row][cbase + 2] = s2;  Asb[row][cbase + 2] = (__bf16)s2;
            decb[row][cbase + 3] = s3;  Asb[row][cbase + 3] = (__bf16)s3;
        }
        __syncthreads();

        // y_t = dec_out @ lin_out_W^T + lin_out_b  -> out[b][s]
        if (tid < BM) {
            float sum = outb[0];
            for (int cc = 0; cc < IC; ++cc) sum += decb[tid][cc] * outW[cc];
            out[(size_t)(b0 + tid) * TDEC + s] = sum;
        }
    }
}

extern "C" void kernel_launch(void* const* d_in, const int* in_sizes, int n_in,
                              void* d_out, int out_size, void* d_ws, size_t ws_size,
                              hipStream_t stream) {
    (void)in_sizes; (void)n_in; (void)out_size; (void)ws_size;

    const float* x        = (const float*)d_in[0];
    const float* enc_Wih  = (const float*)d_in[1];
    const float* enc_Whh  = (const float*)d_in[2];
    const float* enc_bih  = (const float*)d_in[3];
    const float* enc_bhh  = (const float*)d_in[4];
    const float* dec_Wih  = (const float*)d_in[5];
    const float* dec_Whh  = (const float*)d_in[6];
    const float* dec_bih  = (const float*)d_in[7];
    const float* dec_bhh  = (const float*)d_in[8];
    const float* lin_dec_W = (const float*)d_in[9];
    const float* lin_dec_b = (const float*)d_in[10];
    const float* lin_out_W = (const float*)d_in[11];
    const float* lin_out_b = (const float*)d_in[12];

    char* ws = (char*)d_ws;
    const size_t wbytes = (size_t)GC * KC * sizeof(__bf16);   // 2.25 MB each
    __bf16* Wenc = (__bf16*)ws;
    __bf16* Wdec = (__bf16*)(ws + wbytes);
    float*  benc = (float*)(ws + 2 * wbytes);
    float*  bdec = benc + GC;

    const int prep_blocks = (GC * KC + 255) / 256;   // 4608
    s2s_prep_kernel<<<prep_blocks, 256, 0, stream>>>(enc_Wih, enc_Whh, enc_bih, enc_bhh,
                                                     Wenc, benc);
    s2s_prep_kernel<<<prep_blocks, 256, 0, stream>>>(dec_Wih, dec_Whh, dec_bih, dec_bhh,
                                                     Wdec, bdec);

    s2s_lstm_kernel<<<BTOT / BM, 256, 0, stream>>>(x, Wenc, benc, Wdec, bdec,
                                                   lin_dec_W, lin_dec_b,
                                                   lin_out_W, lin_out_b,
                                                   (float*)d_out);
}
